// GatedDeltaNet_10479720202316
// MI455X (gfx1250) — compile-verified, matrix operand order checked by reading, not run on hardware
//
#include <hip/hip_runtime.h>
#include <hip/hip_bf16.h>

// ---------------------------------------------------------------------------
// GatedDeltaNet forward for MI455X (gfx1250, wave32, WMMA)
//   B=4, T=2048, C=1024, H=16, D=64
// Pipeline:
//   1) convert x, Wq/Wk/Wv/Wgate/Wo to bf16
//   2) bf16 WMMA GEMMs (16x16x32): Q,K,V, gate=sigmoid(xWg^T)
//   3) beta = sigmoid(x Wbeta^T + b)        (small N=16 kernel)
//   4) L2-normalize Q,K per head
//   5) chunked delta-rule scan (L=64) with f32 WMMA 16x16x4,
//      async global->LDS staging (ASYNCcnt), fused gate*o -> bf16
//   6) bf16 WMMA GEMM: out = (gate*o) Wo^T  (f32 output)
// ---------------------------------------------------------------------------

#define BB 4
#define TT 2048
#define CC 1024
#define HH 16
#define DD 64
#define MROWS (BB * TT)        // 8192
#define LCH 64                 // scan chunk length
#define NCH (TT / LCH)         // 32 chunks
#define LPAD 68                // LDS row stride: bank-conflict free AND 16B-aligned rows

typedef __attribute__((ext_vector_type(2)))  float  v2f;
typedef __attribute__((ext_vector_type(8)))  float  v8f;
typedef __attribute__((ext_vector_type(16))) __bf16 v16bf;
typedef __attribute__((ext_vector_type(4)))  int    v4i_;

union U16 { v16bf v; uint4 u[2]; };

#if defined(__has_builtin)
#if __has_builtin(__builtin_amdgcn_global_load_async_to_lds_b128) && \
    __has_builtin(__builtin_amdgcn_s_wait_asynccnt)
#define USE_ASYNC_LDS 1
#endif
#endif
#ifndef USE_ASYNC_LDS
#define USE_ASYNC_LDS 0
#endif

#define GPTR(p) ((__attribute__((address_space(1))) v4i_*)(p))
#define LPTR(p) ((__attribute__((address_space(3))) v4i_*)(p))

__device__ __forceinline__ float sigmoidf_(float x) { return 1.0f / (1.0f + __expf(-x)); }

__device__ __forceinline__ v8f wmma_f32x4(v2f a, v2f b, v8f c) {
  return __builtin_amdgcn_wmma_f32_16x16x4_f32(false, a, false, b, (short)0, c, false, false);
}
__device__ __forceinline__ v8f wmma_bf16(v16bf a, v16bf b, v8f c) {
  return __builtin_amdgcn_wmma_f32_16x16x32_bf16(false, a, false, b, (short)0, c, false, false);
}

// ---------------------------------------------------------------------------
// f32 -> bf16 elementwise convert
// ---------------------------------------------------------------------------
__global__ void cvt_bf16_kernel(const float* __restrict__ in,
                                __hip_bfloat16* __restrict__ out, int n) {
  int i = blockIdx.x * blockDim.x + threadIdx.x;
  if (i < n) out[i] = __float2bfloat16(in[i]);
}

// ---------------------------------------------------------------------------
// Y[M,N] = A[M,K] @ Bm[N,K]^T  (bf16 in, f32 accumulate/out), act=1 -> sigmoid
// One wave computes a 16x64 strip (4 adjacent 16x16 tiles): the A fragment is
// reused 4x per K-step, cutting L2 traffic per WMMA by ~40%.
// bf16 A-fragment layout (ISA 7.12.2): lane<16 holds K {0..7,16..23} of row
// M=lane; lane>=16 holds K {8..15,24..31}; B mirrored with N across lanes.
// ---------------------------------------------------------------------------
__global__ __launch_bounds__(256)
void gemm_bf16_nt_kernel(const __hip_bfloat16* __restrict__ A,
                         const __hip_bfloat16* __restrict__ Bm,
                         float* __restrict__ Y,
                         int M, int N, int K, int act) {
  const int wave = threadIdx.x >> 5;
  const int lane = threadIdx.x & 31;
  const int wtile = blockIdx.x * 8 + wave;      // 16 rows x 64 cols each
  const int nt4 = N >> 6;
  const int tm = wtile / nt4;
  const int n0 = (wtile - tm * nt4) * 64;
  if (tm >= (M >> 4)) return;
  const int hi = lane >> 4, r = lane & 15;
  const __hip_bfloat16* arow = A  + (size_t)(tm * 16 + r) * K;
  const __hip_bfloat16* brow = Bm + (size_t)(n0 + r) * K;
  v8f c0 = {}, c1 = {}, c2 = {}, c3 = {};
  for (int k0 = 0; k0 < K; k0 += 32) {
    const int kb = k0 + hi * 8;
    U16 af, b0, b1, b2, b3;
    af.u[0] = *(const uint4*)(arow + kb);
    af.u[1] = *(const uint4*)(arow + kb + 16);
    b0.u[0] = *(const uint4*)(brow + kb);
    b0.u[1] = *(const uint4*)(brow + kb + 16);
    b1.u[0] = *(const uint4*)(brow + 16 * (size_t)K + kb);
    b1.u[1] = *(const uint4*)(brow + 16 * (size_t)K + kb + 16);
    b2.u[0] = *(const uint4*)(brow + 32 * (size_t)K + kb);
    b2.u[1] = *(const uint4*)(brow + 32 * (size_t)K + kb + 16);
    b3.u[0] = *(const uint4*)(brow + 48 * (size_t)K + kb);
    b3.u[1] = *(const uint4*)(brow + 48 * (size_t)K + kb + 16);
    c0 = wmma_bf16(af.v, b0.v, c0);
    c1 = wmma_bf16(af.v, b1.v, c1);
    c2 = wmma_bf16(af.v, b2.v, c2);
    c3 = wmma_bf16(af.v, b3.v, c3);
  }
  float* ybase = Y + (size_t)(tm * 16 + hi * 8) * N + n0 + r;
#pragma unroll
  for (int j = 0; j < 4; ++j) {
    const v8f c = (j == 0) ? c0 : (j == 1) ? c1 : (j == 2) ? c2 : c3;
    float* yb = ybase + j * 16;
#pragma unroll
    for (int rr = 0; rr < 8; ++rr) {
      float v = c[rr];
      if (act == 1) v = sigmoidf_(v);
      yb[(size_t)rr * N] = v;
    }
  }
}

// ---------------------------------------------------------------------------
// beta = sigmoid(x @ Wbeta^T + bbeta); one block per row, f32 throughout.
// ---------------------------------------------------------------------------
__global__ __launch_bounds__(256)
void beta_kernel(const float* __restrict__ X, const float* __restrict__ Wb,
                 const float* __restrict__ bb, float* __restrict__ Beta) {
  __shared__ float sx[CC];
  __shared__ float part[HH][17];
  const int row = blockIdx.x;
  const float* xr = X + (size_t)row * CC;
  for (int i = threadIdx.x; i < CC; i += 256) sx[i] = xr[i];
  __syncthreads();
  const int h = threadIdx.x >> 4, p = threadIdx.x & 15;
  const float* wr = Wb + (size_t)h * CC + p * 64;
  const float* xp = sx + p * 64;
  float acc = 0.f;
#pragma unroll 8
  for (int j = 0; j < 64; ++j) acc += xp[j] * wr[j];
  part[h][p] = acc;
  __syncthreads();
  if (threadIdx.x < HH) {
    float s = bb[threadIdx.x];
#pragma unroll
    for (int p2 = 0; p2 < 16; ++p2) s += part[threadIdx.x][p2];
    Beta[(size_t)row * HH + threadIdx.x] = sigmoidf_(s);
  }
}

// ---------------------------------------------------------------------------
// In-place per-head L2 normalize: one thread per (row, head), 64 floats each.
// ---------------------------------------------------------------------------
__global__ __launch_bounds__(256)
void l2norm_kernel(float* __restrict__ P) {
  const int idx = blockIdx.x * blockDim.x + threadIdx.x;  // (row, head)
  float* p = P + (size_t)(idx >> 4) * CC + (idx & 15) * DD;
  float4 v[16];
  float ss = 0.f;
#pragma unroll
  for (int i = 0; i < 16; ++i) {
    v[i] = *(const float4*)(p + 4 * i);
    ss += v[i].x * v[i].x + v[i].y * v[i].y + v[i].z * v[i].z + v[i].w * v[i].w;
  }
  const float inv = 1.0f / fmaxf(sqrtf(ss), 1e-12f);
#pragma unroll
  for (int i = 0; i < 16; ++i) {
    float4 w = v[i];
    w.x *= inv; w.y *= inv; w.z *= inv; w.w *= inv;
    *(float4*)(p + 4 * i) = w;
  }
}

// ---------------------------------------------------------------------------
// Chunked delta-rule scan. One block (8 waves) per (b,h). State M (64x64 f32)
// lives in LDS across the 32 sequential chunks. Exact chunk algebra:
//   S = Qc Kc^T; A = Kc Kc^T; O0 = Qc M^T; W0 = Vc - Kc M^T
//   U = (I + stril(diag(beta) A))^{-1} diag(beta) W0   (forward substitution)
//   O = O0 + stril(S) U;   M += U^T Kc
// Epilogue: GO = bf16(gate * O)
// Chunk staging uses GLOBAL_LOAD_ASYNC_TO_LDS_B128 (ASYNCcnt) when available.
// f32 WMMA 16x16x4 fragment layout (ISA 7.12.2):
//   A: lane<16 row M=lane, VGPR e -> K = 2*hi + e ; B: lane<16 col N=lane.
// ---------------------------------------------------------------------------
__global__ __launch_bounds__(256)
void scan_kernel(const float* __restrict__ Qg, const float* __restrict__ Kg,
                 const float* __restrict__ Vg, const float* __restrict__ Betag,
                 const float* __restrict__ Gateg, __hip_bfloat16* __restrict__ GOg) {
  __shared__ float sQ[LCH * LPAD], sK[LCH * LPAD], sV[LCH * LPAD],
                   sS[LCH * LPAD], sA[LCH * LPAD], sO[LCH * LPAD],
                   sM[DD * LPAD];
  __shared__ float sb[LCH];
  const int bh = blockIdx.x, b = bh >> 4, h = bh & 15;
  const int tid = threadIdx.x, wave = tid >> 5, lane = tid & 31;
  const int hi = lane >> 4, r = lane & 15;
  const int lr = tid >> 4;         // 0..15 (load row within 16-row slab)
  const int lc = (tid & 15) * 4;   // 0..60 (load col *4)

  for (int i = tid; i < DD * LPAD; i += 256) sM[i] = 0.f;
  __syncthreads();

  const size_t rowbase = (size_t)b * TT;

  for (int ch = 0; ch < NCH; ++ch) {
    const int t0 = ch * LCH;
    // ---- stage chunk: Qc, Kc, Vc (64x64 f32 each) + beta
#if USE_ASYNC_LDS
    for (int rr = 0; rr < LCH; rr += 16) {
      const int row = rr + lr;
      const size_t g = (rowbase + t0 + row) * CC + h * DD + lc;
      const int o = row * LPAD + lc;
      __builtin_amdgcn_global_load_async_to_lds_b128(GPTR(Qg + g), LPTR(&sQ[o]), 0, 0);
      __builtin_amdgcn_global_load_async_to_lds_b128(GPTR(Kg + g), LPTR(&sK[o]), 0, 0);
      __builtin_amdgcn_global_load_async_to_lds_b128(GPTR(Vg + g), LPTR(&sV[o]), 0, 0);
      __builtin_prefetch(Gateg + g, 0, 1);   // gate consumed in the epilogue
    }
    if (tid < LCH) sb[tid] = Betag[(rowbase + t0 + tid) * HH + h];
    __builtin_amdgcn_s_wait_asynccnt(0);
    __syncthreads();
#else
    for (int rr = 0; rr < LCH; rr += 16) {
      const int row = rr + lr;
      const size_t g = (rowbase + t0 + row) * CC + h * DD + lc;
      const float4 q4 = *(const float4*)(Qg + g);
      const float4 k4 = *(const float4*)(Kg + g);
      const float4 v4 = *(const float4*)(Vg + g);
      const int o = row * LPAD + lc;
      *(float4*)&sQ[o] = q4;
      *(float4*)&sK[o] = k4;
      *(float4*)&sV[o] = v4;
      __builtin_prefetch(Gateg + g, 0, 1);
    }
    if (tid < LCH) sb[tid] = Betag[(rowbase + t0 + tid) * HH + h];
    __syncthreads();
#endif

    // ---- P2: 64 tiles over 8 waves:
    //   mm=0: S=Q*K^T  mm=1: A=K*K^T  mm=2: O0=Q*M^T  mm=3: W0 = V - K*M^T
    // All use "transposed-B" addressing: B(k,n) = Bb[n][k].
#pragma unroll 1
    for (int q = 0; q < 8; ++q) {
      const int id = wave * 8 + q;
      const int mm = id >> 4, ti = (id >> 2) & 3, tj = id & 3;
      const float* Ab = (mm == 0 || mm == 2) ? sQ : sK;
      const float* Bb = (mm <= 1) ? sK : sM;
      const int ar = (ti * 16 + r) * LPAD;
      const int br = (tj * 16 + r) * LPAD;
      v8f c = {};
#pragma unroll 4
      for (int k0 = 0; k0 < 64; k0 += 4) {
        const int ka = k0 + 2 * hi;
        v2f a, bv;
        a[0] = Ab[ar + ka];  a[1] = Ab[ar + ka + 1];
        bv[0] = Bb[br + ka]; bv[1] = Bb[br + ka + 1];
        c = wmma_f32x4(a, bv, c);
      }
      const int row0 = ti * 16 + hi * 8, col = tj * 16 + r;
      if (mm == 0) {
#pragma unroll
        for (int rr = 0; rr < 8; ++rr) sS[(row0 + rr) * LPAD + col] = c[rr];
      } else if (mm == 1) {
#pragma unroll
        for (int rr = 0; rr < 8; ++rr) sA[(row0 + rr) * LPAD + col] = c[rr];
      } else if (mm == 2) {
#pragma unroll
        for (int rr = 0; rr < 8; ++rr) sO[(row0 + rr) * LPAD + col] = c[rr];
      } else {
#pragma unroll
        for (int rr = 0; rr < 8; ++rr) sV[(row0 + rr) * LPAD + col] -= c[rr];
      }
    }
    __syncthreads();

    // ---- P3: forward substitution, U overwrites sV
    {
      const int d = tid & 63, grp = tid >> 6;
      for (int rs = 0; rs < LCH; ++rs) {
        if (tid < LCH) sV[rs * LPAD + tid] *= sb[rs];   // finalize u_rs
        __syncthreads();
        const float ur = sV[rs * LPAD + d];
        for (int s = rs + 1 + grp; s < LCH; s += 4)
          sV[s * LPAD + d] -= sA[s * LPAD + rs] * ur;
        __syncthreads();
      }
    }

    // ---- P4: 32 tiles over 8 waves:
    //   mm=0: O += stril(S) * U   mm=1: M += U^T * K
#pragma unroll 1
    for (int q = 0; q < 4; ++q) {
      const int id = wave * 4 + q;
      const int mm = id >> 4, ti = (id >> 2) & 3, tj = id & 3;
      const int row0 = ti * 16 + hi * 8, col = tj * 16 + r;
      v8f c = {};
      if (mm == 0) {
        const int trow = ti * 16 + r;
        const int arow = trow * LPAD;
#pragma unroll 4
        for (int k0 = 0; k0 < 64; k0 += 4) {
          const int ka = k0 + 2 * hi;
          v2f a, bv;
          a[0] = (ka < trow)     ? sS[arow + ka]     : 0.f;   // strict lower mask
          a[1] = (ka + 1 < trow) ? sS[arow + ka + 1] : 0.f;
          bv[0] = sV[ka * LPAD + tj * 16 + r];
          bv[1] = sV[(ka + 1) * LPAD + tj * 16 + r];
          c = wmma_f32x4(a, bv, c);
        }
#pragma unroll
        for (int rr = 0; rr < 8; ++rr) sO[(row0 + rr) * LPAD + col] += c[rr];
      } else {
#pragma unroll 4
        for (int k0 = 0; k0 < 64; k0 += 4) {
          const int ka = k0 + 2 * hi;
          v2f a, bv;
          a[0] = sV[ka * LPAD + ti * 16 + r];        // U^T(d,s) = U[s][d]
          a[1] = sV[(ka + 1) * LPAD + ti * 16 + r];
          bv[0] = sK[ka * LPAD + tj * 16 + r];
          bv[1] = sK[(ka + 1) * LPAD + tj * 16 + r];
          c = wmma_f32x4(a, bv, c);
        }
#pragma unroll
        for (int rr = 0; rr < 8; ++rr) sM[(row0 + rr) * LPAD + col] += c[rr];
      }
    }
    __syncthreads();

    // ---- P5: GO = bf16(gate * O)
    for (int rr = 0; rr < LCH; rr += 16) {
      const int row = rr + lr;
      const size_t g = (rowbase + t0 + row) * CC + h * DD + lc;
      const float4 g4 = *(const float4*)(Gateg + g);
      const int o = row * LPAD + lc;
      GOg[g]     = __float2bfloat16(g4.x * sO[o]);
      GOg[g + 1] = __float2bfloat16(g4.y * sO[o + 1]);
      GOg[g + 2] = __float2bfloat16(g4.z * sO[o + 2]);
      GOg[g + 3] = __float2bfloat16(g4.w * sO[o + 3]);
    }
    __syncthreads();
  }
}

// ---------------------------------------------------------------------------
extern "C" void kernel_launch(void* const* d_in, const int* in_sizes, int n_in,
                              void* d_out, int out_size, void* d_ws, size_t ws_size,
                              hipStream_t stream) {
  (void)in_sizes; (void)n_in; (void)out_size; (void)ws_size;
  const float* x     = (const float*)d_in[0];
  const float* Wq    = (const float*)d_in[1];
  const float* Wk    = (const float*)d_in[2];
  const float* Wv    = (const float*)d_in[3];
  const float* Wbeta = (const float*)d_in[4];
  const float* bbeta = (const float*)d_in[5];
  const float* Wgate = (const float*)d_in[6];
  const float* Wo    = (const float*)d_in[7];
  float* out = (float*)d_out;

  char* ws = (char*)d_ws;
  size_t off = 0;
  auto alloc = [&](size_t bytes) -> void* {
    void* p = ws + off;
    off += (bytes + 255) & ~(size_t)255;
    return p;
  };
  __hip_bfloat16* Xh  = (__hip_bfloat16*)alloc((size_t)MROWS * CC * 2);
  __hip_bfloat16* Whq = (__hip_bfloat16*)alloc((size_t)CC * CC * 2);
  __hip_bfloat16* Whk = (__hip_bfloat16*)alloc((size_t)CC * CC * 2);
  __hip_bfloat16* Whv = (__hip_bfloat16*)alloc((size_t)CC * CC * 2);
  __hip_bfloat16* Whg = (__hip_bfloat16*)alloc((size_t)CC * CC * 2);
  __hip_bfloat16* Who = (__hip_bfloat16*)alloc((size_t)CC * CC * 2);
  float* Qf = (float*)alloc((size_t)MROWS * CC * 4);
  float* Kf = (float*)alloc((size_t)MROWS * CC * 4);
  float* Vf = (float*)alloc((size_t)MROWS * CC * 4);
  float* Gt = (float*)alloc((size_t)MROWS * CC * 4);
  float* Bt = (float*)alloc((size_t)MROWS * HH * 4);
  __hip_bfloat16* GOh = (__hip_bfloat16*)alloc((size_t)MROWS * CC * 2);

  // 1) bf16 conversions
  cvt_bf16_kernel<<<(MROWS * CC) / 256, 256, 0, stream>>>(x, Xh, MROWS * CC);
  cvt_bf16_kernel<<<(CC * CC) / 256, 256, 0, stream>>>(Wq, Whq, CC * CC);
  cvt_bf16_kernel<<<(CC * CC) / 256, 256, 0, stream>>>(Wk, Whk, CC * CC);
  cvt_bf16_kernel<<<(CC * CC) / 256, 256, 0, stream>>>(Wv, Whv, CC * CC);
  cvt_bf16_kernel<<<(CC * CC) / 256, 256, 0, stream>>>(Wgate, Whg, CC * CC);
  cvt_bf16_kernel<<<(CC * CC) / 256, 256, 0, stream>>>(Wo, Who, CC * CC);

  // 2) projections (bf16 WMMA, f32 accumulate); each wave: 16x64 strip
  const int gblocks = (MROWS / 16) * (CC / 64) / 8;  // 1024
  gemm_bf16_nt_kernel<<<gblocks, 256, 0, stream>>>(Xh, Whq, Qf, MROWS, CC, CC, 0);
  gemm_bf16_nt_kernel<<<gblocks, 256, 0, stream>>>(Xh, Whk, Kf, MROWS, CC, CC, 0);
  gemm_bf16_nt_kernel<<<gblocks, 256, 0, stream>>>(Xh, Whv, Vf, MROWS, CC, CC, 0);
  gemm_bf16_nt_kernel<<<gblocks, 256, 0, stream>>>(Xh, Whg, Gt, MROWS, CC, CC, 1);

  // 3) beta (f32)
  beta_kernel<<<MROWS, 256, 0, stream>>>(x, Wbeta, bbeta, Bt);

  // 4) per-head L2 normalize of Q, K (in place)
  l2norm_kernel<<<(MROWS * HH) / 256, 256, 0, stream>>>(Qf);
  l2norm_kernel<<<(MROWS * HH) / 256, 256, 0, stream>>>(Kf);

  // 5) chunked delta-rule scan with fused gating -> bf16
  scan_kernel<<<BB * HH, 256, 0, stream>>>(Qf, Kf, Vf, Bt, Gt, GOh);

  // 6) output projection
  gemm_bf16_nt_kernel<<<gblocks, 256, 0, stream>>>(GOh, Who, out, MROWS, CC, CC, 0);
}
